// AxialMambaBlock_57698590655014
// MI455X (gfx1250) — matrix operand order repaired and checked by reading, not compiled, not run on hardware
//
#include <hip/hip_runtime.h>
#include <stdint.h>

#define LSEQ 56
#define NSEQ 112

typedef __attribute__((ext_vector_type(16))) __bf16 v16bf;
typedef __attribute__((ext_vector_type(8)))  float  v8f;

union FragBF { v16bf v; uint32_t u[8]; };

__device__ inline v8f wmma_bf16(v16bf a, v16bf b, v8f c) {
  return __builtin_amdgcn_wmma_f32_16x16x32_bf16(false, a, false, b, (short)0, c, false, false);
}

// fast hardware approximation paths (v_rcp_f32 / v_rsq_f32 / v_log_f32)
__device__ inline float fast_rcp(float x)  { return __builtin_amdgcn_rcpf(x); }
__device__ inline float fast_rsq(float x)  { return __builtin_amdgcn_rsqf(x); }
__device__ inline float fast_log(float x)  { return __builtin_amdgcn_logf(x) * 0.6931471805599453f; }

// RNE f32 -> bf16 pair pack (lo = even-K element, hi = odd-K element)
__device__ inline uint32_t pack_bf16(float a, float b) {
  uint32_t ua = __builtin_bit_cast(uint32_t, a);
  uint32_t ub = __builtin_bit_cast(uint32_t, b);
  ua = (ua + 0x7fffu + ((ua >> 16) & 1u)) >> 16;
  ub = (ub + 0x7fffu + ((ub >> 16) & 1u)) >> 16;
  return (ub << 16) | (ua & 0xffffu);
}

// A-matrix 16x32 bf16 fragment (ISA 7.12.2): lane row=l&15; VGPR v holds K pair
// kpair = (v<4 ? v : v+4) + (l>=16 ? 4 : 0). Source: packed pairs, row-major [M][ld].
__device__ inline v16bf load_a(const uint32_t* p, int ld, int mbase, int kp16, int lane) {
  int hi = lane >> 4, row = lane & 15;
  FragBF f;
#pragma unroll
  for (int v = 0; v < 8; ++v) {
    int kp = (v < 4 ? v : v + 4) + hi * 4 + kp16;
    f.u[v] = p[(mbase + row) * ld + kp];
  }
  return f.v;
}

// B-matrix 32x16 bf16 fragment: lane col=l&15; kpair = (v<4 ? v : v+4) + (l>=16 ? 4 : 0).
// Source: packed pairs [Kp2][Npad].
__device__ inline v16bf load_b(const uint32_t* p, int ld, int kp16, int nbase, int lane) {
  int hi = lane >> 4, col = lane & 15;
  FragBF f;
#pragma unroll
  for (int v = 0; v < 8; ++v) {
    int kp = kp16 + (v < 4 ? v : v + 4) + hi * 4;
    f.u[v] = p[kp * ld + nbase + col];
  }
  return f.v;
}

// C/D 16x16 f32: lane l, VGPR v -> (M = v + (l>=16?8:0), N = l&15)
__device__ inline void store_d(float* p, int ld, int mbase, int nbase, int lane, v8f acc) {
  int hi = lane >> 4, col = lane & 15;
#pragma unroll
  for (int v = 0; v < 8; ++v) p[(mbase + v + hi * 8) * ld + nbase + col] = acc[v];
}

__device__ inline float silu(float a) { return a * fast_rcp(1.f + __expf(-a)); }

// ---------------------------------------------------------------------------
// K0a: pack weight [N][K] f32 (row-major, out_features x in_features) into
// bf16 pair layout dst[kp][n] for WMMA B operand; zero-pad n>=N.
__global__ void pack_weights_kernel(const float* __restrict__ src, uint32_t* __restrict__ dst,
                                    int N, int K, int Npad) {
  int total = (K / 2) * Npad;
  for (int i = blockIdx.x * blockDim.x + threadIdx.x; i < total; i += gridDim.x * blockDim.x) {
    int kp = i / Npad, n = i % Npad;
    float a = (n < N) ? src[n * K + 2 * kp] : 0.f;
    float b = (n < N) ? src[n * K + 2 * kp + 1] : 0.f;
    dst[i] = pack_bf16(a, b);
  }
}

// K0b: A = clip(-exp(clip(A_log,-5,5)), -5, 5)  (upper clip is a no-op)
__global__ void prep_A_kernel(const float* __restrict__ A_log, float* __restrict__ Aneg) {
  int i = blockIdx.x * blockDim.x + threadIdx.x;
  if (i < 192 * 96) {
    float a = -__expf(fminf(fmaxf(A_log[i], -5.f), 5.f));
    Aneg[i] = fmaxf(a, -5.f);
  }
}

// K1: down-project x (b,64,h,w) by dw (96,64) into per-sequence token rows,
// packed bf16 pairs [s][64 rows (pad>=56 zero)][48 kpairs] for WMMA A operand.
// dir=0: s=b*56+w, t=h.  dir=1: s=b*56+h, t=w.
__global__ void seq_prep_kernel(const float* __restrict__ x, const float* __restrict__ dw,
                                uint32_t* __restrict__ seq_pk, int dir) {
  int total = NSEQ * 64 * 48;
  for (int i = blockIdx.x * blockDim.x + threadIdx.x; i < total; i += gridDim.x * blockDim.x) {
    int cp = i % 48;
    int t  = (i / 48) & 63;
    int s  = i / (48 * 64);
    int b  = s / 56, q = s % 56;
    float a0 = 0.f, a1 = 0.f;
    if (t < LSEQ) {
      int h  = dir ? q : t;
      int wi = dir ? t : q;
      const float* xp = x + ((size_t)(b * 64) * 56 + h) * 56 + wi;
      const float* w0 = dw + (2 * cp) * 64;
      const float* w1 = w0 + 64;
      __builtin_prefetch(xp, 0, 0);           // global_prefetch_b8
#pragma unroll 4
      for (int k = 0; k < 64; ++k) {
        float xv = xp[(size_t)k * 3136];
        a0 += xv * w0[k];
        a1 += xv * w1[k];
      }
    }
    seq_pk[i] = pack_bf16(a0, a1);
  }
}

// ---------------------------------------------------------------------------
// K2: fused mamba block, one workgroup per sequence, 8 wave32s.
// LDS layout (f32 elements), ~306 KB of the 320 KB/WGP budget:
//   XZ  [64][384]        0      (in_proj output; [:,192:] = res gate, kept live)
//   UF  [56][192]        24576  (silu(conv) u -> later overwritten with g = dt*u)
//   UPK [64][96] u32     35328  (u bf16 pairs -> later y2 bf16 pairs)
//   XD  [64][208]        41472  (x_dbl; cols 0..5 dt-in, 6..101 B, 102..197 C)
//   DT  [56][192]        54784  (softplus delta -> in-place inclusive prefix Pd)
//   TD  [192]            65536  (total delta sum per channel)
//   YA  [56][192]        65728  (y accumulator, init u*D)
__global__ __launch_bounds__(256)
void mamba_kernel(const uint32_t* __restrict__ seq_pk,
                  const uint32_t* __restrict__ inW,     // [48][384] pairs
                  const uint32_t* __restrict__ xprojW,  // [96][208] pairs
                  const uint32_t* __restrict__ outW,    // [96][96]  pairs
                  const float* __restrict__ Aneg,       // [192][96]
                  const float* __restrict__ conv_w,     // [192][4]
                  const float* __restrict__ conv_b,     // [192]
                  const float* __restrict__ dproj_w,    // [192][6]
                  const float* __restrict__ dproj_b,    // [192]
                  const float* __restrict__ Dp,         // [192]
                  float* __restrict__ out_mx) {         // [s][56][96]
  extern __shared__ float sm[];
  float*    XZ  = sm;
  float*    UF  = sm + 24576;
  uint32_t* UPK = (uint32_t*)(sm + 35328);
  float*    XD  = sm + 41472;
  float*    DT  = sm + 54784;
  float*    TD  = sm + 65536;
  float*    YA  = sm + 65728;

  const int tid  = threadIdx.x;
  const int lane = tid & 31;
  const int wv   = tid >> 5;
  const int s    = blockIdx.x;
  const uint32_t* A0 = seq_pk + (size_t)s * 64 * 48;

  // ---- GEMM1: xz[64x384] = seq(64x96) x in_w^T, bf16 WMMA, K=96 (3 steps) ----
  for (int tile = wv; tile < 96; tile += 8) {
    int mt = (tile & 3) * 16, nt = (tile >> 2) * 16;
    v8f acc = {};
#pragma unroll
    for (int ks = 0; ks < 3; ++ks) {
      v16bf a = load_a(A0, 48, mt, ks * 16, lane);
      v16bf b = load_b(inW, 384, ks * 16, nt, lane);
      acc = wmma_bf16(a, b, acc);
    }
    store_d(XZ, 384, mt, nt, lane, acc);
  }
  __syncthreads();

  // ---- causal depthwise conv (K=4) + SiLU; store u f32 + bf16 pairs ----
  for (int i = tid; i < 64 * 96; i += 256) {
    int t = i / 96, dp = i % 96;
    float u0 = 0.f, u1 = 0.f;
    if (t < LSEQ) {
#pragma unroll
      for (int h = 0; h < 2; ++h) {
        int d = 2 * dp + h;
        float acc = conv_b[d];
#pragma unroll
        for (int j = 0; j < 4; ++j) {
          int tt = t - 3 + j;
          if (tt >= 0) acc += XZ[tt * 384 + d] * conv_w[d * 4 + j];
        }
        float uv = silu(acc);
        if (h == 0) u0 = uv; else u1 = uv;
      }
      UF[t * 192 + 2 * dp]     = u0;
      UF[t * 192 + 2 * dp + 1] = u1;
    }
    UPK[t * 96 + dp] = pack_bf16(u0, u1);
  }
  __syncthreads();

  // ---- GEMM2: x_dbl[64x208] = u(64x192) x xproj_w^T, K=192 (6 steps) ----
  for (int tile = wv; tile < 52; tile += 8) {
    int mt = (tile % 4) * 16, nt = (tile / 4) * 16;
    v8f acc = {};
#pragma unroll
    for (int ks = 0; ks < 6; ++ks) {
      v16bf a = load_a(UPK, 96, mt, ks * 16, lane);
      v16bf b = load_b(xprojW, 208, ks * 16, nt, lane);
      acc = wmma_bf16(a, b, acc);
    }
    store_d(XD, 208, mt, nt, lane, acc);
  }
  __syncthreads();

  // ---- delta = softplus(x_dbl[:,:6] @ dproj^T + b); g = dt*u; YA = u*D ----
  for (int i = tid; i < LSEQ * 192; i += 256) {
    int t = i / 192, d = i % 192;
    float acc = dproj_b[d];
#pragma unroll
    for (int j = 0; j < 6; ++j) acc += XD[t * 208 + j] * dproj_w[d * 6 + j];
    float dt = (acc > 20.f) ? acc : fast_log(1.f + __expf(acc));
    DT[i] = dt;
    float u = UF[i];
    YA[i] = u * Dp[d];
    UF[i] = dt * u;           // g = delta * u
  }
  __syncthreads();

  // ---- inclusive prefix of delta over t, per channel (in-place) ----
  if (tid < 192) {
    float acc = 0.f;
    for (int t = 0; t < LSEQ; ++t) { acc += DT[t * 192 + tid]; DT[t * 192 + tid] = acc; }
    TD[tid] = acc;
  }
  __syncthreads();

  // ---- selective scan: thread d, 16-state register chunks over n ----
  // c[t,n] = exp(min(A_n*(Td - P[t]), 15));  num += g*B*c;  y += C*num*rcp(c+1e-6)
  if (tid < 192) {
    const int d = tid;
    const float td = TD[d];
    for (int nb = 0; nb < 6; ++nb) {
      float Av[16], num[16];
#pragma unroll
      for (int j = 0; j < 16; ++j) { Av[j] = Aneg[d * 96 + nb * 16 + j]; num[j] = 0.f; }
      for (int t = 0; t < LSEQ; ++t) {
        float gd = UF[t * 192 + d];
        float Sd = td - DT[t * 192 + d];
        float contrib = 0.f;
#pragma unroll
        for (int j = 0; j < 16; ++j) {
          float c = __expf(fminf(Av[j] * Sd, 15.f));
          num[j] += gd * XD[t * 208 + 6 + nb * 16 + j] * c;
          contrib += XD[t * 208 + 102 + nb * 16 + j] * num[j] * fast_rcp(c + 1e-6f);
        }
        YA[t * 192 + d] += contrib;
      }
    }
  }
  __syncthreads();

  // ---- gate: y2 = y * silu(res); repack bf16 pairs into UPK ----
  for (int i = tid; i < 64 * 96; i += 256) {
    int t = i / 96, dp = i % 96;
    float y0 = 0.f, y1 = 0.f;
    if (t < LSEQ) {
      y0 = YA[t * 192 + 2 * dp]     * silu(XZ[t * 384 + 192 + 2 * dp]);
      y1 = YA[t * 192 + 2 * dp + 1] * silu(XZ[t * 384 + 192 + 2 * dp + 1]);
    }
    UPK[t * 96 + dp] = pack_bf16(y0, y1);
  }
  __syncthreads();

  // ---- GEMM3: out[56x96] = y2(64x192) x out_w^T, K=192 (6 steps) -> global ----
  float* out_s = out_mx + (size_t)s * LSEQ * 96;
  for (int tile = wv; tile < 24; tile += 8) {
    int mt = (tile % 4) * 16, nt = (tile / 4) * 16;
    v8f acc = {};
#pragma unroll
    for (int ks = 0; ks < 6; ++ks) {
      v16bf a = load_a(UPK, 96, mt, ks * 16, lane);
      v16bf b = load_b(outW, 96, ks * 16, nt, lane);
      acc = wmma_bf16(a, b, acc);
    }
    int hi = lane >> 4, col = lane & 15;
#pragma unroll
    for (int v = 0; v < 8; ++v) {
      int row = mt + v + hi * 8;
      if (row < LSEQ) out_s[row * 96 + nt + col] = acc[v];
    }
  }
}

// K3: up-project both directions (K=96 each), batchnorm affine, residual add.
__global__ void combine_kernel(const float* __restrict__ x,
                               const float* __restrict__ hmx, const float* __restrict__ wmx,
                               const float* __restrict__ hu_w, const float* __restrict__ wu_w,
                               const float* __restrict__ hg, const float* __restrict__ hb,
                               const float* __restrict__ hmean, const float* __restrict__ hvar,
                               const float* __restrict__ wg, const float* __restrict__ wb,
                               const float* __restrict__ wmean, const float* __restrict__ wvar,
                               float* __restrict__ out) {
  int total = 2 * 64 * 56 * 56;
  for (int i = blockIdx.x * blockDim.x + threadIdx.x; i < total; i += gridDim.x * blockDim.x) {
    int w = i % 56;
    int h = (i / 56) % 56;
    int o = (i / 3136) % 64;
    int b = i / 200704;
    const float* hp = hmx + ((size_t)(b * 56 + w) * 56 + h) * 96;
    const float* wp = wmx + ((size_t)(b * 56 + h) * 56 + w) * 96;
    const float* hw = hu_w + o * 96;
    const float* ww = wu_w + o * 96;
    float hs = 0.f, wsum = 0.f;
#pragma unroll 4
    for (int d = 0; d < 96; ++d) { hs += hp[d] * hw[d]; wsum += wp[d] * ww[d]; }
    float hinv = hg[o] * fast_rsq(hvar[o] + 1e-5f);   // v_rsq_f32
    float winv = wg[o] * fast_rsq(wvar[o] + 1e-5f);
    out[i] = (hs * hinv + hb[o] - hmean[o] * hinv)
           + (wsum * winv + wb[o] - wmean[o] * winv) + x[i];
  }
}

// ---------------------------------------------------------------------------
extern "C" void kernel_launch(void* const* d_in, const int* in_sizes, int n_in,
                              void* d_out, int out_size, void* d_ws, size_t ws_size,
                              hipStream_t stream) {
  (void)in_sizes; (void)n_in; (void)out_size; (void)ws_size;
  // setup_inputs() dict order:
  const float* x        = (const float*)d_in[0];
  const float* hd_w     = (const float*)d_in[1];
  const float* hu_w     = (const float*)d_in[2];
  const float* wd_w     = (const float*)d_in[3];
  const float* wu_w     = (const float*)d_in[4];
  const float* hn_g     = (const float*)d_in[5];
  const float* hn_b     = (const float*)d_in[6];
  const float* hn_m     = (const float*)d_in[7];
  const float* hn_v     = (const float*)d_in[8];
  const float* wn_g     = (const float*)d_in[9];
  const float* wn_b     = (const float*)d_in[10];
  const float* wn_m     = (const float*)d_in[11];
  const float* wn_v     = (const float*)d_in[12];
  const float* hm_in_w  = (const float*)d_in[13];
  const float* hm_cw    = (const float*)d_in[14];
  const float* hm_cb    = (const float*)d_in[15];
  const float* hm_xp    = (const float*)d_in[16];
  const float* hm_dw    = (const float*)d_in[17];
  const float* hm_db    = (const float*)d_in[18];
  const float* hm_Alog  = (const float*)d_in[19];
  const float* hm_D     = (const float*)d_in[20];
  const float* hm_ow    = (const float*)d_in[21];
  const float* wm_in_w  = (const float*)d_in[22];
  const float* wm_cw    = (const float*)d_in[23];
  const float* wm_cb    = (const float*)d_in[24];
  const float* wm_xp    = (const float*)d_in[25];
  const float* wm_dw    = (const float*)d_in[26];
  const float* wm_db    = (const float*)d_in[27];
  const float* wm_Alog  = (const float*)d_in[28];
  const float* wm_D     = (const float*)d_in[29];
  const float* wm_ow    = (const float*)d_in[30];
  float* out = (float*)d_out;

  // workspace layout (4-byte elements)
  float*    wsf = (float*)d_ws;
  uint32_t* wsu = (uint32_t*)d_ws;
  size_t o = 0;
  uint32_t* hm_inW = wsu + o; o += 48 * 384;
  uint32_t* wm_inW = wsu + o; o += 48 * 384;
  uint32_t* hm_xpW = wsu + o; o += 96 * 208;
  uint32_t* wm_xpW = wsu + o; o += 96 * 208;
  uint32_t* hm_oW  = wsu + o; o += 96 * 96;
  uint32_t* wm_oW  = wsu + o; o += 96 * 96;
  float*    hm_A   = wsf + o; o += 192 * 96;
  float*    wm_A   = wsf + o; o += 192 * 96;
  uint32_t* hseq   = wsu + o; o += NSEQ * 64 * 48;
  uint32_t* wseq   = wsu + o; o += NSEQ * 64 * 48;
  float*    hmx    = wsf + o; o += NSEQ * 56 * 96;
  float*    wmx    = wsf + o; o += NSEQ * 56 * 96;

  auto grid = [](int n) { return (n + 255) / 256; };

  pack_weights_kernel<<<grid(48 * 384), 256, 0, stream>>>(hm_in_w, hm_inW, 384, 96, 384);
  pack_weights_kernel<<<grid(48 * 384), 256, 0, stream>>>(wm_in_w, wm_inW, 384, 96, 384);
  pack_weights_kernel<<<grid(96 * 208), 256, 0, stream>>>(hm_xp, hm_xpW, 198, 192, 208);
  pack_weights_kernel<<<grid(96 * 208), 256, 0, stream>>>(wm_xp, wm_xpW, 198, 192, 208);
  pack_weights_kernel<<<grid(96 * 96), 256, 0, stream>>>(hm_ow, hm_oW, 96, 192, 96);
  pack_weights_kernel<<<grid(96 * 96), 256, 0, stream>>>(wm_ow, wm_oW, 96, 192, 96);
  prep_A_kernel<<<grid(192 * 96), 256, 0, stream>>>(hm_Alog, hm_A);
  prep_A_kernel<<<grid(192 * 96), 256, 0, stream>>>(wm_Alog, wm_A);

  seq_prep_kernel<<<grid(NSEQ * 64 * 48), 256, 0, stream>>>(x, hd_w, hseq, 0);
  seq_prep_kernel<<<grid(NSEQ * 64 * 48), 256, 0, stream>>>(x, wd_w, wseq, 1);

  const size_t smem = 76480 * sizeof(float);   // ~306 KB of 320 KB/WGP
  mamba_kernel<<<NSEQ, 256, smem, stream>>>(hseq, hm_inW, hm_xpW, hm_oW, hm_A,
                                            hm_cw, hm_cb, hm_dw, hm_db, hm_D, hmx);
  mamba_kernel<<<NSEQ, 256, smem, stream>>>(wseq, wm_inW, wm_xpW, wm_oW, wm_A,
                                            wm_cw, wm_cb, wm_dw, wm_db, wm_D, wmx);

  combine_kernel<<<grid(2 * 64 * 56 * 56), 256, 0, stream>>>(
      x, hmx, wmx, hu_w, wu_w, hn_g, hn_b, hn_m, hn_v, wn_g, wn_b, wn_m, wn_v, out);
}
